// CrossAttentionLayer_72636486910047
// MI455X (gfx1250) — matrix operand order repaired and checked by reading, not compiled, not run on hardware
//
#include <hip/hip_runtime.h>
#include <stdint.h>

// ---------------- types ----------------
typedef __bf16 bf16;
typedef __attribute__((ext_vector_type(16))) __bf16 v16bf;
typedef __attribute__((ext_vector_type(8)))  __bf16 v8bf;
typedef __attribute__((ext_vector_type(8)))  float  v8f;
typedef __attribute__((ext_vector_type(4)))  float  v4f;

// Problem constants (from reference setup_inputs)
constexpr int kB = 8;
constexpr int kN = 2048;   // query rows
constexpr int kM = 2048;   // context rows
constexpr int kD = 1024;   // feature dim

__device__ __forceinline__ bf16 f2bf(float x) {
    unsigned u = __builtin_bit_cast(unsigned, x);
    u += 0x7FFFu + ((u >> 16) & 1u);            // round-to-nearest-even
    unsigned short h = (unsigned short)(u >> 16);
    return __builtin_bit_cast(bf16, h);
}

// 16-bit A-fragment for v_wmma_*_16x16x32: lane holds two contiguous
// 8-element K chunks (VGPR0-3 and VGPR4-7) per ISA 7.12.2.
__device__ __forceinline__ v16bf make_frag(const bf16* p0, const bf16* p1) {
    v8bf lo = *(const v8bf*)p0;
    v8bf hi = *(const v8bf*)p1;
    return __builtin_shufflevector(lo, hi, 0,1,2,3,4,5,6,7,8,9,10,11,12,13,14,15);
}

__device__ __forceinline__ v8f wmma_bf16(v16bf a, v16bf b, v8f c) {
    return __builtin_amdgcn_wmma_f32_16x16x32_bf16(false, a, false, b, (short)0, c,
                                                   false, false);
}

// generic (shared) pointer -> 32-bit LDS byte offset (low 32 bits of the
// shared-aperture generic address are the LDS offset)
__device__ __forceinline__ unsigned lds_off(const void* p) {
    return (unsigned)(uintptr_t)p;
}

// ====================================================================
// Kernel 1: fused Q/K/V projection.  y = x @ W.T + bias, output bf16.
// which = 0 -> Q (row-major), 1 -> K (row-major), 2 -> V stored transposed
// (Vt[b][e][m]) so the PV GEMM B-fragments are contiguous in K=m.
// 256 threads / 8 waves, 128x128 tile, K-panels of 32, double-buffered LDS.
// ====================================================================
__global__ __launch_bounds__(256)
void qkv_proj_kernel(const float* __restrict__ query,
                     const float* __restrict__ context,
                     const float* __restrict__ Wq, const float* __restrict__ bq,
                     const float* __restrict__ Wk, const float* __restrict__ bk,
                     const float* __restrict__ Wv, const float* __restrict__ bv,
                     bf16* __restrict__ Qb, bf16* __restrict__ Kb,
                     bf16* __restrict__ Vt) {
    __shared__ bf16 lA[2][128 * 32];
    __shared__ bf16 lB[2][128 * 32];

    const int tid  = threadIdx.x;
    const int lane = tid & 31;
    const int wave = tid >> 5;
    const int half = lane >> 4;
    const int l15  = lane & 15;

    const int z = blockIdx.z;
    const int b = z / 3;
    const int which = z % 3;
    const int rowT = blockIdx.y * 128;
    const int eT   = blockIdx.x * 128;

    const float* X    = (which == 0) ? query : context;
    const float* W    = (which == 0) ? Wq : ((which == 1) ? Wk : Wv);
    const float* bias = (which == 0) ? bq : ((which == 1) ? bk : bv);

    // staging map: each thread converts 16 f32 -> 16 bf16 per panel
    const int sr = tid >> 1;           // 0..127 tile row
    const int sc = (tid & 1) * 16;     // 0 or 16
    const float* gA = X + ((size_t)b * kM + rowT + sr) * kD + sc;
    const float* gW = W + (size_t)(eT + sr) * kD + sc;

    const int waveM = wave & 3;        // 4 waves along rows (32 each)
    const int waveN = wave >> 2;       // 2 waves along cols (64 each)

    auto stage = [&](int buf, int ks) {
        const float* pa = gA + ks * 32;
        const float* pw = gW + ks * 32;
        v16bf va, vb;
        #pragma unroll
        for (int i = 0; i < 4; ++i) {
            v4f fa = *(const v4f*)(pa + i * 4);
            v4f fw = *(const v4f*)(pw + i * 4);
            #pragma unroll
            for (int j = 0; j < 4; ++j) {
                va[i * 4 + j] = f2bf(fa[j]);
                vb[i * 4 + j] = f2bf(fw[j]);
            }
        }
        *(v16bf*)&lA[buf][sr * 32 + sc] = va;
        *(v16bf*)&lB[buf][sr * 32 + sc] = vb;
    };

    v8f acc[2][4];
    #pragma unroll
    for (int mt = 0; mt < 2; ++mt)
        #pragma unroll
        for (int nt = 0; nt < 4; ++nt)
            #pragma unroll
            for (int r = 0; r < 8; ++r) acc[mt][nt][r] = 0.0f;

    stage(0, 0);
    __syncthreads();

    for (int ks = 0; ks < 32; ++ks) {
        const int cur = ks & 1;
        if (ks + 1 < 32) stage(cur ^ 1, ks + 1);   // overlap with WMMAs below

        v16bf af[2], bfm[4];
        #pragma unroll
        for (int mt = 0; mt < 2; ++mt) {
            const int rr = waveM * 32 + mt * 16 + l15;
            const int c0 = half * 16;
            af[mt] = make_frag(&lA[cur][rr * 32 + c0], &lA[cur][rr * 32 + c0 + 8]);
        }
        #pragma unroll
        for (int nt = 0; nt < 4; ++nt) {
            const int cc = waveN * 64 + nt * 16 + l15;
            bfm[nt] = *(const v16bf*)&lB[cur][cc * 32 + half * 16];
        }
        #pragma unroll
        for (int mt = 0; mt < 2; ++mt)
            #pragma unroll
            for (int nt = 0; nt < 4; ++nt)
                acc[mt][nt] = wmma_bf16(af[mt], bfm[nt], acc[mt][nt]);
        __syncthreads();
    }

    // ---- epilogue: branch-free scatter (strides picked once) ----
    bf16* outp;
    size_t rs, cs;
    if (which == 0)      { outp = Qb + (size_t)b * kN * kD; rs = kD; cs = 1;  }
    else if (which == 1) { outp = Kb + (size_t)b * kM * kD; rs = kD; cs = 1;  }
    else                 { outp = Vt + (size_t)b * kD * kM; rs = 1;  cs = kM; }

    #pragma unroll
    for (int mt = 0; mt < 2; ++mt) {
        const int rowbase = rowT + waveM * 32 + mt * 16 + half * 8;
        #pragma unroll
        for (int nt = 0; nt < 4; ++nt) {
            const int col = eT + waveN * 64 + nt * 16 + l15;
            const float bb = bias[col];
            bf16* pc = outp + (size_t)col * cs + (size_t)rowbase * rs;
            #pragma unroll
            for (int r = 0; r < 8; ++r)
                pc[(size_t)r * rs] = f2bf(acc[mt][nt][r] + bb);
        }
    }
}

// ====================================================================
// Kernel 2: flash attention with online softmax.
// Workgroup: (b, 32 query rows), 8 waves. m-tiles of 128.
// Q block (32x1024 bf16, 64KB) staged ONCE into LDS via
// global_load_async_to_lds_b128 (ASYNCcnt) -> kills the 8x redundant
// per-wave global Q fragment traffic of the naive version.
// ====================================================================
__global__ __launch_bounds__(256)
void attn_kernel(const bf16* __restrict__ Qb, const bf16* __restrict__ Kb,
                 const bf16* __restrict__ Vt, const int* __restrict__ cmask,
                 float* __restrict__ out) {
    extern __shared__ char smem[];
    bf16* lQ = (bf16*)smem;                         // 32 x 1024 (64KB)
    bf16* lP = (bf16*)(smem + 32 * 1024 * 2);       // 32 x 128  (8KB)

    __shared__ float lRed[8][32];
    __shared__ float lMrun[32], lLrun[32], lAlpha[32], lMnew[32];

    const int tid  = threadIdx.x;
    const int lane = tid & 31;
    const int wave = tid >> 5;
    const int half = lane >> 4;
    const int l15  = lane & 15;

    const int b  = blockIdx.y;
    const int n0 = blockIdx.x * 32;
    const float scale = 0.03125f;                   // 1/sqrt(1024)

    // ---- async-stage the whole Q block into LDS (256B per thread) ----
    {
        const bf16* gq = Qb + ((size_t)b * kN + n0) * kD;   // 64KB block
        uint64_t gsrc = (uint64_t)(uintptr_t)gq + (uint64_t)tid * 256u;
        unsigned dst  = lds_off(lQ) + (unsigned)tid * 256u;
        #pragma unroll
        for (int j = 0; j < 16; ++j) {
            asm volatile("global_load_async_to_lds_b128 %0, %1, off offset:%2"
                         :: "v"(dst), "v"(gsrc), "i"(j * 16) : "memory");
        }
        asm volatile("s_wait_asynccnt 0x0" ::: "memory");
    }
    if (tid < 32) { lMrun[tid] = -3.0e38f; lLrun[tid] = 0.0f; }

    v8f accO[2][8];
    #pragma unroll
    for (int mt = 0; mt < 2; ++mt)
        #pragma unroll
        for (int nt = 0; nt < 8; ++nt)
            #pragma unroll
            for (int r = 0; r < 8; ++r) accO[mt][nt][r] = 0.0f;
    __syncthreads();   // Q staged + stats initialized

    for (int it = 0; it < 16; ++it) {
        const int m0     = it * 128;
        const int cmbase = m0 + wave * 16;          // this wave's score columns

        // prefetch next m-tile's K/V rows for our lanes
        if (it + 1 < 16) {
            __builtin_prefetch(Kb + ((size_t)b * kM + cmbase + 128 + l15) * kD, 0, 1);
            __builtin_prefetch(Vt + ((size_t)b * kD + wave * 128 + l15) * kM + m0 + 128, 0, 1);
        }

        // ---------- S = Q @ K^T for our 32x16 slice (K from global, Q from LDS) ----------
        v8f accS[2];
        #pragma unroll
        for (int mt = 0; mt < 2; ++mt)
            #pragma unroll
            for (int r = 0; r < 8; ++r) accS[mt][r] = 0.0f;

        for (int ks = 0; ks < 32; ++ks) {
            const int dbase = ks * 32;
            v16bf aq[2];
            #pragma unroll
            for (int mt = 0; mt < 2; ++mt) {
                const bf16* pq = lQ + (mt * 16 + l15) * kD + dbase + half * 16;
                aq[mt] = make_frag(pq, pq + 8);
            }
            const bf16* pk = Kb + ((size_t)b * kM + cmbase + l15) * kD
                                + dbase + half * 16;
            const v16bf bk_ = *(const v16bf*)pk;
            accS[0] = wmma_bf16(aq[0], bk_, accS[0]);
            accS[1] = wmma_bf16(aq[1], bk_, accS[1]);
        }

        // ---------- scale + mask ----------
        const int mv = cmask[(size_t)b * kM + cmbase + l15];
        float sv[2][8];
        #pragma unroll
        for (int mt = 0; mt < 2; ++mt)
            #pragma unroll
            for (int r = 0; r < 8; ++r) {
                float s = accS[mt][r] * scale;
                sv[mt][r] = (mv == 0) ? -1.0e9f : s;
            }

        // ---------- per-wave row max (reduce across 16 lanes) ----------
        #pragma unroll
        for (int mt = 0; mt < 2; ++mt)
            #pragma unroll
            for (int r = 0; r < 8; ++r) {
                float v = sv[mt][r];
                v = fmaxf(v, __shfl_xor(v, 1, 32));
                v = fmaxf(v, __shfl_xor(v, 2, 32));
                v = fmaxf(v, __shfl_xor(v, 4, 32));
                v = fmaxf(v, __shfl_xor(v, 8, 32));
                if (l15 == 0) lRed[wave][mt * 16 + half * 8 + r] = v;
            }
        __syncthreads();

        // ---------- combine across waves, online-softmax stats ----------
        if (tid < 32) {
            float tmax = lRed[0][tid];
            #pragma unroll
            for (int w = 1; w < 8; ++w) tmax = fmaxf(tmax, lRed[w][tid]);
            const float mnew = fmaxf(lMrun[tid], tmax);
            lAlpha[tid] = __expf(lMrun[tid] - mnew);
            lMnew[tid]  = mnew;
            lMrun[tid]  = mnew;
        }
        __syncthreads();

        // ---------- exp, write P to LDS, per-wave row sums ----------
        #pragma unroll
        for (int mt = 0; mt < 2; ++mt)
            #pragma unroll
            for (int r = 0; r < 8; ++r) {
                const int row = mt * 16 + half * 8 + r;
                const float p = __expf(sv[mt][r] - lMnew[row]);
                lP[row * 128 + wave * 16 + l15] = f2bf(p);
                float v = p;
                v += __shfl_xor(v, 1, 32);
                v += __shfl_xor(v, 2, 32);
                v += __shfl_xor(v, 4, 32);
                v += __shfl_xor(v, 8, 32);
                if (l15 == 0) lRed[wave][row] = v;
            }
        __syncthreads();

        if (tid < 32) {
            float ts = 0.0f;
            #pragma unroll
            for (int w = 0; w < 8; ++w) ts += lRed[w][tid];
            lLrun[tid] = lLrun[tid] * lAlpha[tid] + ts;
        }

        // ---------- rescale O by alpha, then O += P @ V ----------
        #pragma unroll
        for (int mt = 0; mt < 2; ++mt)
            #pragma unroll
            for (int r = 0; r < 8; ++r) {
                const float a = lAlpha[mt * 16 + half * 8 + r];
                #pragma unroll
                for (int nt = 0; nt < 8; ++nt) accO[mt][nt][r] *= a;
            }

        #pragma unroll
        for (int ks2 = 0; ks2 < 4; ++ks2) {
            v16bf ap[2];
            #pragma unroll
            for (int mt = 0; mt < 2; ++mt) {
                const bf16* pp = &lP[(mt * 16 + l15) * 128 + ks2 * 32 + half * 16];
                ap[mt] = make_frag(pp, pp + 8);
            }
            #pragma unroll
            for (int nt = 0; nt < 8; ++nt) {
                const int e = wave * 128 + nt * 16 + l15;
                const bf16* pv = Vt + ((size_t)b * kD + e) * kM
                                    + m0 + ks2 * 32 + half * 16;
                const v16bf bv_ = *(const v16bf*)pv;
                accO[0][nt] = wmma_bf16(ap[0], bv_, accO[0][nt]);
                accO[1][nt] = wmma_bf16(ap[1], bv_, accO[1][nt]);
            }
        }
        __syncthreads();
    }

    // ---------- normalize and write f32 output ----------
    #pragma unroll
    for (int mt = 0; mt < 2; ++mt)
        #pragma unroll
        for (int r = 0; r < 8; ++r) {
            const int row = mt * 16 + half * 8 + r;
            const float inv = 1.0f / lLrun[row];
            const int row_g = n0 + row;
            #pragma unroll
            for (int nt = 0; nt < 8; ++nt) {
                const int col = wave * 128 + nt * 16 + l15;
                out[((size_t)b * kN + row_g) * kD + col] = accO[mt][nt][r] * inv;
            }
        }
}

// ====================================================================
extern "C" void kernel_launch(void* const* d_in, const int* in_sizes, int n_in,
                              void* d_out, int out_size, void* d_ws, size_t ws_size,
                              hipStream_t stream) {
    const float* query   = (const float*)d_in[0];
    const float* context = (const float*)d_in[1];
    const int*   cmask   = (const int*)d_in[2];
    const float* Wq = (const float*)d_in[3];
    const float* bq = (const float*)d_in[4];
    const float* Wk = (const float*)d_in[5];
    const float* bk = (const float*)d_in[6];
    const float* Wv = (const float*)d_in[7];
    const float* bv = (const float*)d_in[8];
    float* out = (float*)d_out;

    // workspace: Q (bf16 [b][n][d]) | K (bf16 [b][m][d]) | Vt (bf16 [b][e][m])
    bf16* Qb = (bf16*)d_ws;
    bf16* Kb = Qb + (size_t)kB * kN * kD;
    bf16* Vt = Kb + (size_t)kB * kM * kD;

    dim3 g1(kD / 128, kN / 128, kB * 3);
    qkv_proj_kernel<<<g1, 256, 0, stream>>>(query, context, Wq, bq, Wk, bk, Wv, bv,
                                            Qb, Kb, Vt);

    const size_t smem = (size_t)(32 * 1024 + 32 * 128) * sizeof(bf16);  // 72KB
    dim3 g2(kN / 32, kB);
    attn_kernel<<<g2, 256, smem, stream>>>(Qb, Kb, Vt, cmask, out);
}
